// LinearAttention_45878840656615
// MI455X (gfx1250) — matrix operand order repaired and checked by reading, not compiled, not run on hardware
//
#include <hip/hip_runtime.h>
#include <hip/hip_bf16.h>
#include <math.h>

// ---------------- types for WMMA ----------------
typedef __attribute__((ext_vector_type(16))) __bf16 v16bf;
typedef __attribute__((ext_vector_type(8)))  float  v8f;

// ---------------- problem constants ----------------
#define D_MODEL   1024
#define NH        16
#define DH        64
#define B_BATCH   4
#define T_SEQ     4096
#define M_TOTAL   (B_BATCH * T_SEQ)        // 16384
#define EPSF      1e-6f
#define NCHUNK    32
#define CHUNK_LEN (T_SEQ / NCHUNK)         // 128

// output layout (floats): out [B,T,D], then kv_acc [B,H,64,64], then z_acc [B,H,64]
#define OUT_KV_OFF 16777216
#define OUT_Z_OFF  17039360

// ---------------- helpers ----------------
__device__ __forceinline__ unsigned short f2bf(float x) {
    unsigned int u = __float_as_uint(x);
    unsigned int r = u + 0x7fffu + ((u >> 16) & 1u);   // round-to-nearest-even
    return (unsigned short)(r >> 16);
}

// fast softplus: v_exp_f32 + v_log_f32 instead of a ~50-op inlined log1pf
__device__ __forceinline__ float softplusf(float x) {
    float ax = fabsf(x);
    float l  = __logf(1.f + __expf(-ax));
    return (x > 0.f) ? x + l : l;
}

// LDS byte address of a shared-memory pointer (flat aperture: low 32 bits)
__device__ __forceinline__ unsigned lds_addr(const void* p) {
    return (unsigned)(unsigned long long)p;
}

// GLOBAL_LOAD_ASYNC_TO_LDS_B128 (GVS mode): LDS[vdst_lds] = MEM[saddr + vgoff]
__device__ __forceinline__ void async_b128(unsigned lds_off, unsigned goff,
                                           unsigned long long base) {
    asm volatile("global_load_async_to_lds_b128 %0, %1, %2"
                 :
                 : "v"(lds_off), "v"(goff), "s"(base)
                 : "memory");
}

__device__ __forceinline__ void wait_async_le3() {
    asm volatile("s_wait_asynccnt 0x3" ::: "memory");
}
__device__ __forceinline__ void wait_async_0() {
    asm volatile("s_wait_asynccnt 0x0" ::: "memory");
}

// ---------------- fp32 -> bf16 convert ----------------
__global__ void convert_f32_bf16(const float* __restrict__ in,
                                 unsigned short* __restrict__ out, int n) {
    int i = blockIdx.x * blockDim.x + threadIdx.x;
    int stride = gridDim.x * blockDim.x;
    for (; i < n; i += stride) out[i] = f2bf(in[i]);
}

// ---------------- WMMA bf16 GEMM:  C[m,n] = sum_k A[m,k] * W[n,k]  ----------------
// Block tile 128(M) x 64(N), K staged in LDS by 32 (= WMMA K), double-buffered
// with GLOBAL_LOAD_ASYNC_TO_LDS_B128.  8 waves: wave (wm 0..3, wn 0..1)
// computes a 32x32 sub-tile = 2x2 WMMA tiles.  K-loop unrolled by 2 with the
// final stage pair peeled so buffer indices are compile-time constants.
enum { EPI_V = 0, EPI_QK = 1, EPI_OUT = 2 };

#define A_STRIDE 40
#define A_BUFSZ  (128 * A_STRIDE)   // elems
#define W_BUFSZ  (64 * A_STRIDE)    // elems
#define KSTAGES  (D_MODEL / 32)     // 32

template <int EPI>
__global__ __launch_bounds__(256)
void gemm_wmma_bf16(const __bf16* __restrict__ A,
                    const __bf16* __restrict__ W,
                    float* __restrict__ outp) {
    __shared__ __align__(16) __bf16 As[2][A_BUFSZ];
    __shared__ __align__(16) __bf16 Ws[2][W_BUFSZ];

    const int tid  = threadIdx.x;
    const int lane = tid & 31;
    const int wave = tid >> 5;
    const int wm   = wave >> 1;          // 0..3
    const int wn   = wave & 1;           // 0..1
    const int half = lane >> 4;          // 0: lanes 0-15, 1: lanes 16-31
    const int l15  = lane & 15;
    const int m0   = blockIdx.y * 128;
    const int n0   = blockIdx.x * 64;

    // staging geometry: per thread 2 async b128 for A (rows r, r+64), 1 for W
    const int rowA = tid >> 2;           // 0..63
    const int colA = (tid & 3) * 8;      // 0,8,16,24 (elems)
    const unsigned long long baseA = (unsigned long long)A;
    const unsigned long long baseW = (unsigned long long)W;

    // global byte offsets (advance by 64B per stage)
    unsigned gA0 = (unsigned)(((m0 + rowA) * D_MODEL + colA) * 2);
    unsigned gA1 = gA0 + (unsigned)(64 * D_MODEL * 2);
    unsigned gW  = (unsigned)(((n0 + rowA) * D_MODEL + colA) * 2);

    // LDS byte offsets per buffer
    unsigned lA0[2], lA1[2], lW[2];
#pragma unroll
    for (int b = 0; b < 2; ++b) {
        lA0[b] = lds_addr(&As[b][rowA * A_STRIDE + colA]);
        lA1[b] = lA0[b] + (unsigned)(64 * A_STRIDE * 2);
        lW[b]  = lds_addr(&Ws[b][rowA * A_STRIDE + colA]);
    }

    v8f acc[2][2];
#pragma unroll
    for (int i = 0; i < 2; ++i)
#pragma unroll
        for (int j = 0; j < 2; ++j)
#pragma unroll
            for (int r = 0; r < 8; ++r) acc[i][j][r] = 0.f;

    // issue one full stage of 3 async copies into buffer `buf`
    auto issue = [&](int buf) {
        async_b128(lA0[buf], gA0, baseA);
        async_b128(lA1[buf], gA1, baseA);
        async_b128(lW[buf],  gW,  baseW);
    };
    auto advance = [&]() { gA0 += 64; gA1 += 64; gW += 64; };

    // compute one K-step from buffer `cur` (cur is a literal at every callsite)
    auto compute = [&](int cur) {
        // A fragment (16x32): lane m = l15;
        //  lanes 0-15 : elems 0..15 -> K 0..15
        //  lanes16-31 : elems 0..15 -> K 16..31
        v16bf af[2], bfr[2];
#pragma unroll
        for (int i = 0; i < 2; ++i) {
            const int rb = wm * 32 + i * 16;
            const __bf16* ap = &As[cur][(rb + l15) * A_STRIDE + half * 16];
#pragma unroll
            for (int v = 0; v < 8; ++v) {
                af[i][v]     = ap[v];
                af[i][v + 8] = ap[8 + v];
            }
        }
        // B fragment (32x16): lane n = l15; lanes 0-15: K 0..15; lanes 16-31: K 16..31
#pragma unroll
        for (int j = 0; j < 2; ++j) {
            const int cb = wn * 32 + j * 16;
            const __bf16* bp = &Ws[cur][(cb + l15) * A_STRIDE + half * 16];
#pragma unroll
            for (int v = 0; v < 16; ++v) bfr[j][v] = bp[v];
        }
#pragma unroll
        for (int i = 0; i < 2; ++i)
#pragma unroll
            for (int j = 0; j < 2; ++j)
                acc[i][j] = __builtin_amdgcn_wmma_f32_16x16x32_bf16(
                    false, af[i], false, bfr[j], (short)0, acc[i][j], false, false);
    };

    // prologue: stage 0 -> buffer 0
    issue(0);

    // steady state: stages 0..29 (15 double-iterations), always prefetching
    for (int kt = 0; kt < KSTAGES - 2; kt += 2) {
        advance(); issue(1); wait_async_le3();
        __syncthreads();
        compute(0);
        __syncthreads();

        advance(); issue(0); wait_async_le3();
        __syncthreads();
        compute(1);
        __syncthreads();
    }
    // peeled tail: stage 30 in buf0 (prefetch stage 31 into buf1), stage 31 in buf1
    advance(); issue(1); wait_async_le3();
    __syncthreads();
    compute(0);
    __syncthreads();

    wait_async_0();
    __syncthreads();
    compute(1);

    // epilogue: C layout — lanes 0-15: M=r, lanes 16-31: M=8+r; N = l15
#pragma unroll
    for (int i = 0; i < 2; ++i) {
#pragma unroll
        for (int j = 0; j < 2; ++j) {
#pragma unroll
            for (int r = 0; r < 8; ++r) {
                int m = m0 + wm * 32 + i * 16 + (half ? (r + 8) : r);
                int n = n0 + wn * 32 + j * 16 + l15;
                float val = acc[i][j][r];
                if constexpr (EPI == EPI_QK) val = softplusf(val) + EPSF;
                if constexpr (EPI == EPI_OUT) {
                    outp[(size_t)m * D_MODEL + n] = val;
                } else {
                    // scatter [B,H,T,dh]
                    int b = m >> 12, t = m & (T_SEQ - 1);
                    int h = n >> 6,  e = n & 63;
                    outp[(((size_t)(b * NH + h) * T_SEQ + t) << 6) + e] = val;
                }
            }
        }
    }
}

// ---------------- scan pass 1: chunk-local sums of k (x) v and k ----------------
__global__ __launch_bounds__(256)
void scan_pass1(const float* __restrict__ kf, const float* __restrict__ vf,
                float* __restrict__ pkv, float* __restrict__ pz) {
    __shared__ float k_lds[64], v_lds[64];
    const int bid  = blockIdx.x;              // (b*16+h)*NCHUNK + c
    const int c    = bid & (NCHUNK - 1);
    const int bh   = bid >> 5;
    const int tid  = threadIdx.x;
    const int e    = tid & 63;
    const int dgrp = tid >> 6;                // 0..3
    const int t0   = c * CHUNK_LEN;
    const size_t base = ((size_t)bh * T_SEQ + t0) * DH;

    float kv[16];
#pragma unroll
    for (int i = 0; i < 16; ++i) kv[i] = 0.f;
    float zacc = 0.f;

    for (int t = 0; t < CHUNK_LEN; ++t) {
        if (tid < 64)        k_lds[tid]      = kf[base + (size_t)t * DH + tid];
        else if (tid < 128)  v_lds[tid - 64] = vf[base + (size_t)t * DH + (tid - 64)];
        __syncthreads();
        float vv = v_lds[e];
#pragma unroll
        for (int i = 0; i < 16; ++i) kv[i] += k_lds[dgrp * 16 + i] * vv;
        if (tid < 64) zacc += k_lds[tid];
        __syncthreads();
    }
    const size_t pb = (size_t)bid * 4096;
#pragma unroll
    for (int i = 0; i < 16; ++i) pkv[pb + (dgrp * 16 + i) * 64 + e] = kv[i];
    if (tid < 64) pz[(size_t)bid * 64 + tid] = zacc;
}

// ---------------- scan pass 2: exclusive prefix over chunks; emit kv_acc / z_acc ----------------
__global__ __launch_bounds__(256)
void scan_pass2(float* __restrict__ pkv, float* __restrict__ pz,
                float* __restrict__ kv_out, float* __restrict__ z_out) {
    const int bh  = blockIdx.x;               // 0..63
    const int tid = threadIdx.x;
    for (int idx = tid; idx < 4096; idx += 256) {
        float run = 0.f;
        for (int c = 0; c < NCHUNK; ++c) {
            size_t off = ((size_t)(bh * NCHUNK + c)) * 4096 + idx;
            float tv = pkv[off];
            pkv[off] = run;
            run += tv;
        }
        kv_out[(size_t)bh * 4096 + idx] = run;
    }
    if (tid < 64) {
        float run = 0.f;
        for (int c = 0; c < NCHUNK; ++c) {
            size_t off = ((size_t)(bh * NCHUNK + c)) * 64 + tid;
            float tv = pz[off];
            pz[off] = run;
            run += tv;
        }
        z_out[bh * 64 + tid] = run;
    }
}

// ---------------- scan pass 3: replay chunk from prefix, emit per-t outputs (bf16) ----------------
__global__ __launch_bounds__(256)
void scan_pass3(const float* __restrict__ qf, const float* __restrict__ kf,
                const float* __restrict__ vf, const float* __restrict__ pkv,
                const float* __restrict__ pz, unsigned short* __restrict__ attn) {
    __shared__ float q_lds[64], k_lds[64], v_lds[64];
    __shared__ float red[256];
    __shared__ float qz[64];
    const int bid  = blockIdx.x;
    const int c    = bid & (NCHUNK - 1);
    const int bh   = bid >> 5;
    const int b    = bh >> 4, h = bh & 15;
    const int tid  = threadIdx.x;
    const int e    = tid & 63;
    const int dgrp = tid >> 6;
    const int t0   = c * CHUNK_LEN;
    const size_t base = ((size_t)bh * T_SEQ + t0) * DH;

    float kv[16];
    const size_t pb = (size_t)bid * 4096;
#pragma unroll
    for (int i = 0; i < 16; ++i) kv[i] = pkv[pb + (dgrp * 16 + i) * 64 + e];
    float zreg = (tid < 64) ? pz[(size_t)bid * 64 + tid] : 0.f;

    for (int t = 0; t < CHUNK_LEN; ++t) {
        if (tid < 64)        q_lds[tid]       = qf[base + (size_t)t * DH + tid];
        else if (tid < 128)  k_lds[tid - 64]  = kf[base + (size_t)t * DH + (tid - 64)];
        else if (tid < 192)  v_lds[tid - 128] = vf[base + (size_t)t * DH + (tid - 128)];
        __syncthreads();

        float vv = v_lds[e];
        float p = 0.f;
#pragma unroll
        for (int i = 0; i < 16; ++i) {
            kv[i] += k_lds[dgrp * 16 + i] * vv;   // kv += k (x) v  (before use, as in ref)
            p     += q_lds[dgrp * 16 + i] * kv[i];
        }
        red[dgrp * 64 + e] = p;
        if (tid < 64) {
            zreg += k_lds[tid];                   // z += k (before den, as in ref)
            qz[tid] = q_lds[tid] * zreg;
        }
        __syncthreads();

        if (tid < 64) {
            float num = red[tid] + red[64 + tid] + red[128 + tid] + red[192 + tid];
            float den = EPSF;
#pragma unroll 8
            for (int d = 0; d < 64; ++d) den += qz[d];
            int tt = t0 + t;
            attn[((size_t)(b * T_SEQ + tt)) * D_MODEL + h * DH + tid] = f2bf(num / den);
        }
        __syncthreads();
    }
}

// ---------------- host launcher ----------------
extern "C" void kernel_launch(void* const* d_in, const int* in_sizes, int n_in,
                              void* d_out, int out_size, void* d_ws, size_t ws_size,
                              hipStream_t stream) {
    (void)in_sizes; (void)n_in; (void)out_size; (void)ws_size;

    const float* x  = (const float*)d_in[0];
    const float* Wq = (const float*)d_in[1];
    const float* Wk = (const float*)d_in[2];
    const float* Wv = (const float*)d_in[3];
    const float* Wo = (const float*)d_in[4];
    float* out = (float*)d_out;

    char* ws = (char*)d_ws;
    size_t off = 0;
    auto alloc = [&](size_t bytes) -> char* {
        char* p = ws + off;
        off += (bytes + 255) & ~(size_t)255;
        return p;
    };
    unsigned short* xb  = (unsigned short*)alloc((size_t)M_TOTAL * D_MODEL * 2);
    unsigned short* wqb = (unsigned short*)alloc((size_t)D_MODEL * D_MODEL * 2);
    unsigned short* wkb = (unsigned short*)alloc((size_t)D_MODEL * D_MODEL * 2);
    unsigned short* wvb = (unsigned short*)alloc((size_t)D_MODEL * D_MODEL * 2);
    unsigned short* wob = (unsigned short*)alloc((size_t)D_MODEL * D_MODEL * 2);
    float* qf = (float*)alloc((size_t)M_TOTAL * D_MODEL * 4);
    float* kf = (float*)alloc((size_t)M_TOTAL * D_MODEL * 4);
    float* vf = (float*)alloc((size_t)M_TOTAL * D_MODEL * 4);
    unsigned short* attnb = (unsigned short*)alloc((size_t)M_TOTAL * D_MODEL * 2);
    float* pkv = (float*)alloc((size_t)B_BATCH * NH * NCHUNK * 4096 * 4);
    float* pz  = (float*)alloc((size_t)B_BATCH * NH * NCHUNK * 64 * 4);

    // 1) fp32 -> bf16
    convert_f32_bf16<<<4096, 256, 0, stream>>>(x,  xb,  M_TOTAL * D_MODEL);
    convert_f32_bf16<<<1024, 256, 0, stream>>>(Wq, wqb, D_MODEL * D_MODEL);
    convert_f32_bf16<<<1024, 256, 0, stream>>>(Wk, wkb, D_MODEL * D_MODEL);
    convert_f32_bf16<<<1024, 256, 0, stream>>>(Wv, wvb, D_MODEL * D_MODEL);
    convert_f32_bf16<<<1024, 256, 0, stream>>>(Wo, wob, D_MODEL * D_MODEL);

    // 2) projections with fused phi for q/k, scatter to [B,H,T,dh]
    dim3 gg(D_MODEL / 64, M_TOTAL / 128);
    gemm_wmma_bf16<EPI_QK><<<gg, 256, 0, stream>>>((const __bf16*)xb, (const __bf16*)wqb, qf);
    gemm_wmma_bf16<EPI_QK><<<gg, 256, 0, stream>>>((const __bf16*)xb, (const __bf16*)wkb, kf);
    gemm_wmma_bf16<EPI_V ><<<gg, 256, 0, stream>>>((const __bf16*)xb, (const __bf16*)wvb, vf);

    // 3) chunk-parallel linear-attention scan
    scan_pass1<<<B_BATCH * NH * NCHUNK, 256, 0, stream>>>(kf, vf, pkv, pz);
    scan_pass2<<<B_BATCH * NH, 256, 0, stream>>>(pkv, pz, out + OUT_KV_OFF, out + OUT_Z_OFF);
    scan_pass3<<<B_BATCH * NH * NCHUNK, 256, 0, stream>>>(qf, kf, vf, pkv, pz, attnb);

    // 4) output projection -> d_out [B,T,D] fp32
    gemm_wmma_bf16<EPI_OUT><<<gg, 256, 0, stream>>>((const __bf16*)attnb, (const __bf16*)wob, out);
}
